// LSTM2d_81509889343601
// MI455X (gfx1250) — matrix operand order repaired and checked by reading, not compiled, not run on hardware
//
#include <hip/hip_runtime.h>
#include <hip/hip_bf16.h>

// ---------------- types / constants ----------------
typedef unsigned short u16;
typedef __attribute__((ext_vector_type(16))) __bf16 v16bf;
typedef __attribute__((ext_vector_type(8)))  float  v8f;

#define JJ   64      // input length
#define II   64      // output length
#define BB   32      // batch
#define EMB  512
#define ENC  512
#define DD2  512
#define VV   32000
#define NGENC 2048   // 4*ENC encoder gate width
#define NROW  2560   // 5*D2 decoder gate width
#define XIN   1536   // EMBED + 2*ENC

struct U8 { unsigned u[8]; };

static __device__ __forceinline__ u16 f2bf(float x) {
    unsigned u = __builtin_bit_cast(unsigned, x);
    unsigned r = u + 0x7FFFu + ((u >> 16) & 1u);
    return (u16)(r >> 16);
}
static __device__ __forceinline__ float sigf(float x) { return 1.f / (1.f + __expf(-x)); }

static __device__ __forceinline__ v8f wmma_bf16(v16bf a, v16bf b, v8f c) {
    return __builtin_amdgcn_wmma_f32_16x16x32_bf16(false, a, false, b, (short)0, c, false, false);
}
// 16-bit A-matrix 16x32 fragment (ISA 7.12.2): lane<16 holds K {2r}/{16+2r}, lane>=16 +8
static __device__ __forceinline__ int kA(int r, int half) {
    return ((r < 4) ? 2 * r : 16 + 2 * (r - 4)) + 8 * half;
}
// A: row-major (16 rows starting at A), leading dim ldk (elements)
static __device__ __forceinline__ v16bf loadA(const u16* A, size_t ldk, int kbase, int lane) {
    int half = lane >> 4, m = lane & 15;
    U8 f;
#pragma unroll
    for (int r = 0; r < 8; ++r)
        f.u[r] = *(const unsigned*)(A + (size_t)m * ldk + kbase + kA(r, half));
    return __builtin_bit_cast(v16bf, f);
}
// B is W[n,k] row-major (so B^T of the K x N operand); Bbase pre-offset to first col row.
static __device__ __forceinline__ v16bf loadB(const u16* Bbase, size_t ldk, int kbase, int lane) {
    int half = lane >> 4, n = lane & 15;
    U8 f;
#pragma unroll
    for (int r = 0; r < 8; ++r)
        f.u[r] = *(const unsigned*)(Bbase + (size_t)n * ldk + kbase + kA(r, half));
    return __builtin_bit_cast(v16bf, f);
}

// ---------------- setup kernels ----------------
__global__ void k_convert(const float* __restrict__ s, u16* __restrict__ d, size_t n) {
    size_t i = (size_t)blockIdx.x * blockDim.x + threadIdx.x;
    size_t stride = (size_t)gridDim.x * blockDim.x;
    for (; i < n; i += stride) d[i] = f2bf(s[i]);
}

__global__ void k_embed_x(const int* __restrict__ x, const float* __restrict__ emb,
                          u16* __restrict__ embX) {
    size_t i = (size_t)blockIdx.x * blockDim.x + threadIdx.x;   // over J*B*EMB
    if (i >= (size_t)JJ * BB * EMB) return;
    int e = (int)(i % EMB);
    int jb = (int)(i / EMB);
    int tok = x[jb];                                            // x is (J,B) row-major
    embX[i] = f2bf(emb[(size_t)tok * EMB + e]);
}

// ---------------- generic wave-per-tile bf16 GEMM: C = A * W^T (+bias) ----------------
// A:(Mt*16 x KT*32) bf16 lda; W:(Nt*16 x ldb) bf16, using cols [koff, koff+KT*32)
__global__ __launch_bounds__(256) void k_gemm_waves(
    const u16* __restrict__ A, int lda,
    const u16* __restrict__ W, int ldb, int koff,
    const float* __restrict__ bias,
    float* __restrict__ C, int ldc, int Mt, int Nt, int KT) {
    int lane = threadIdx.x & 31;
    int tile = blockIdx.x * (blockDim.x >> 5) + (threadIdx.x >> 5);
    if (tile >= Mt * Nt) return;
    int mt = tile % Mt, nt = tile / Mt;
    const u16* Abase = A + (size_t)(mt * 16) * lda;
    const u16* Bbase = W + (size_t)(nt * 16) * ldb + koff;
    v8f acc = {};
    for (int kt = 0; kt < KT; ++kt) {
        v16bf a = loadA(Abase, lda, kt * 32, lane);
        v16bf b = loadB(Bbase, ldb, kt * 32, lane);
        acc = wmma_bf16(a, b, acc);
    }
    int half = lane >> 4, col = lane & 15;
    int n = nt * 16 + col;
    float bv = bias ? bias[n] : 0.f;
#pragma unroll
    for (int r = 0; r < 8; ++r) {
        int m = mt * 16 + r + 8 * half;
        C[(size_t)m * ldc + n] = acc[r] + bv;
    }
}

// ---------------- encoder: one persistent WG per direction ----------------
// wave w owns units [16w,16w+16) of all 4 gates; c stays in VGPRs, h in LDS bf16.
// Loop nest: kt outer, gates inner -> each A fragment loaded once, used by 4 WMMAs.
__global__ __launch_bounds__(1024) void k_encoder(
    const u16* __restrict__ embX,
    const u16* __restrict__ WihF, const u16* __restrict__ WhhF, const float* __restrict__ bF,
    const u16* __restrict__ WihB, const u16* __restrict__ WhhB, const float* __restrict__ bB,
    u16* __restrict__ henc) {
    __shared__ u16 h_lds[BB * ENC];                       // 32 KB
    const int dir = blockIdx.x;
    const u16* Wih = dir ? WihB : WihF;
    const u16* Whh = dir ? WhhB : WhhF;
    const float* bias = dir ? bB : bF;
    int tid = threadIdx.x, lane = tid & 31, w = tid >> 5;
    int half = lane >> 4, col = lane & 15;
    int u = w * 16 + col;
    for (int idx = tid; idx < BB * ENC; idx += 1024) h_lds[idx] = 0;
    v8f c[2] = {};
    __syncthreads();
    for (int t = 0; t < JJ; ++t) {
        int j = dir ? (JJ - 1 - t) : t;
        const u16* Aemb = embX + (size_t)j * BB * EMB;
        v8f acc[4][2] = {};
#pragma unroll
        for (int mt = 0; mt < 2; ++mt) {
            const u16* Ae = Aemb + (size_t)(mt * 16) * EMB;
            const u16* Ah = h_lds + (size_t)(mt * 16) * ENC;
            for (int kt = 0; kt < 16; ++kt) {
                v16bf aE = loadA(Ae, EMB, kt * 32, lane);   // one A-frag ...
                v16bf aH = loadA(Ah, ENC, kt * 32, lane);
#pragma unroll
                for (int g = 0; g < 4; ++g) {               // ... feeds all 4 gates
                    const u16* Bih = Wih + (size_t)(g * 512 + w * 16) * EMB;
                    const u16* Bhh = Whh + (size_t)(g * 512 + w * 16) * ENC;
                    acc[g][mt] = wmma_bf16(aE, loadB(Bih, EMB, kt * 32, lane), acc[g][mt]);
                    acc[g][mt] = wmma_bf16(aH, loadB(Bhh, ENC, kt * 32, lane), acc[g][mt]);
                }
            }
        }
        __syncthreads();                                  // all waves done reading h_lds
        float bi = bias[u], bf2 = bias[512 + u], bg = bias[1024 + u], bo = bias[1536 + u];
#pragma unroll
        for (int mt = 0; mt < 2; ++mt)
#pragma unroll
            for (int r = 0; r < 8; ++r) {
                int m = mt * 16 + r + 8 * half;
                float gi = acc[0][mt][r] + bi;
                float gf = acc[1][mt][r] + bf2;
                float gg = acc[2][mt][r] + bg;
                float go = acc[3][mt][r] + bo;
                float cc = sigf(gf) * c[mt][r] + sigf(gi) * tanhf(gg);
                c[mt][r] = cc;
                u16 hb = f2bf(sigf(go) * tanhf(cc));
                h_lds[m * ENC + u] = hb;
                henc[((size_t)j * BB + m) * (2 * ENC) + dir * ENC + u] = hb;
            }
        __syncthreads();
    }
}

// ---------------- decoder row scan: one persistent WG ----------------
// pre = Hpart[j] + Ypart + s * Wsh^T ; wave w owns units [16w,16w+16) of all 5 gates.
// kt outer / gates inner: each s-state A fragment loaded from LDS once per 5 WMMAs.
__global__ __launch_bounds__(1024) void k_row(
    const float* __restrict__ Hpart, const float* __restrict__ Ypart,
    const u16* __restrict__ Ws, u16* __restrict__ sJ) {
    __shared__ u16 s_lds[BB * DD2];                       // 32 KB
    int tid = threadIdx.x, lane = tid & 31, w = tid >> 5;
    int half = lane >> 4, col = lane & 15;
    int u = w * 16 + col;
    for (int idx = tid; idx < BB * DD2; idx += 1024) s_lds[idx] = 0;
    v8f c[2] = {};
    __syncthreads();
    for (int j = 0; j < JJ; ++j) {
        v8f acc[5][2];
#pragma unroll
        for (int mt = 0; mt < 2; ++mt) {
            // init accumulators with the hoisted input-dependent parts (+ bias, inside Hpart)
#pragma unroll
            for (int g = 0; g < 5; ++g) {
                int n = g * 512 + u;
#pragma unroll
                for (int r = 0; r < 8; ++r) {
                    int m = mt * 16 + r + 8 * half;
                    acc[g][mt][r] = Hpart[((size_t)j * BB + m) * NROW + n] +
                                    Ypart[(size_t)m * NROW + n];
                }
            }
            const u16* As = s_lds + (size_t)(mt * 16) * DD2;
            for (int kt = 0; kt < 16; ++kt) {             // only s_hor half (s_ver == 0)
                v16bf aS = loadA(As, DD2, kt * 32, lane);
#pragma unroll
                for (int g = 0; g < 5; ++g) {
                    const u16* Bws = Ws + (size_t)(g * 512 + w * 16) * (2 * DD2);
                    acc[g][mt] = wmma_bf16(aS, loadB(Bws, 2 * DD2, kt * 32, lane), acc[g][mt]);
                }
            }
        }
        __syncthreads();
#pragma unroll
        for (int mt = 0; mt < 2; ++mt)
#pragma unroll
            for (int r = 0; r < 8; ++r) {
                int m = mt * 16 + r + 8 * half;
                float ii = sigf(acc[0][mt][r]);
                float ff = sigf(acc[1][mt][r]);
                float oo = sigf(acc[2][mt][r]);
                float ll = sigf(acc[3][mt][r]);
                float gg = tanhf(acc[4][mt][r]);
                float cc = ff * (ll * c[mt][r]) + ii * gg;
                c[mt][r] = cc;
                s_lds[m * DD2 + u] = f2bf(oo * tanhf(cc));
            }
        __syncthreads();
    }
    for (int idx = tid; idx < BB * DD2; idx += 1024) sJ[idx] = s_lds[idx];
}

// ---------------- softmax + argmax + greedy embedding feedback ----------------
__global__ __launch_bounds__(1024) void k_softmax_feedback(
    float* __restrict__ probs, const float* __restrict__ embedding, u16* __restrict__ yprev) {
    __shared__ float rmax[1024];
    __shared__ int   ridx[1024];
    __shared__ float rsum[1024];
    int b = blockIdx.x, tid = threadIdx.x;
    float* row = probs + (size_t)b * VV;
    float mx = -3.4e38f; int mi = 0;
    for (int v = tid; v < VV; v += 1024) { float x = row[v]; if (x > mx) { mx = x; mi = v; } }
    rmax[tid] = mx; ridx[tid] = mi; __syncthreads();
    for (int s = 512; s > 0; s >>= 1) {
        if (tid < s) {
            float o = rmax[tid + s]; int oi = ridx[tid + s];
            if (o > rmax[tid] || (o == rmax[tid] && oi < ridx[tid])) { rmax[tid] = o; ridx[tid] = oi; }
        }
        __syncthreads();
    }
    mx = rmax[0];
    float sum = 0.f;
    for (int v = tid; v < VV; v += 1024) { float e = __expf(row[v] - mx); row[v] = e; sum += e; }
    rsum[tid] = sum; __syncthreads();
    for (int s = 512; s > 0; s >>= 1) { if (tid < s) rsum[tid] += rsum[tid + s]; __syncthreads(); }
    float inv = 1.f / rsum[0];
    for (int v = tid; v < VV; v += 1024) row[v] *= inv;
    int tok = ridx[0];
    if (tid < EMB) yprev[(size_t)b * EMB + tid] = f2bf(embedding[(size_t)tok * EMB + tid]);
}

// ---------------- host ----------------
static inline size_t alignup(size_t x) { return (x + 255) & ~(size_t)255; }

extern "C" void kernel_launch(void* const* d_in, const int* in_sizes, int n_in,
                              void* d_out, int out_size, void* d_ws, size_t ws_size,
                              hipStream_t stream) {
    (void)in_sizes; (void)n_in; (void)out_size; (void)ws_size;
    const int*   x       = (const int*)d_in[0];
    const float* emb     = (const float*)d_in[2];
    const float* WihF    = (const float*)d_in[3];
    const float* WhhF    = (const float*)d_in[4];
    const float* bF      = (const float*)d_in[5];
    const float* WihB    = (const float*)d_in[6];
    const float* WhhB    = (const float*)d_in[7];
    const float* bB      = (const float*)d_in[8];
    const float* Wx      = (const float*)d_in[9];
    const float* Ws      = (const float*)d_in[10];
    const float* b_cell  = (const float*)d_in[11];
    const float* Wl      = (const float*)d_in[12];
    const float* bl      = (const float*)d_in[13];
    float* out = (float*)d_out;

    char* ws = (char*)d_ws;
    size_t off = 0;
    auto alloc = [&](size_t bytes) { char* p = ws + off; off += alignup(bytes); return p; };
    u16* wWihF = (u16*)alloc((size_t)NGENC * EMB * 2);
    u16* wWhhF = (u16*)alloc((size_t)NGENC * ENC * 2);
    u16* wWihB = (u16*)alloc((size_t)NGENC * EMB * 2);
    u16* wWhhB = (u16*)alloc((size_t)NGENC * ENC * 2);
    u16* wWx   = (u16*)alloc((size_t)NROW * XIN * 2);
    u16* wWs   = (u16*)alloc((size_t)NROW * 2 * DD2 * 2);
    u16* wWl   = (u16*)alloc((size_t)VV * DD2 * 2);
    u16* embX  = (u16*)alloc((size_t)JJ * BB * EMB * 2);
    u16* henc  = (u16*)alloc((size_t)JJ * BB * 2 * ENC * 2);
    float* Hpart = (float*)alloc((size_t)JJ * BB * NROW * 4);
    float* Ypart = (float*)alloc((size_t)BB * NROW * 4);
    u16* sJ    = (u16*)alloc((size_t)BB * DD2 * 2);
    u16* yprev = (u16*)alloc((size_t)BB * EMB * 2);

    // weight conversion f32 -> bf16 (all weights then live in L2: ~55 MB of 192 MB)
    k_convert<<<2048, 256, 0, stream>>>(WihF, wWihF, (size_t)NGENC * EMB);
    k_convert<<<2048, 256, 0, stream>>>(WhhF, wWhhF, (size_t)NGENC * ENC);
    k_convert<<<2048, 256, 0, stream>>>(WihB, wWihB, (size_t)NGENC * EMB);
    k_convert<<<2048, 256, 0, stream>>>(WhhB, wWhhB, (size_t)NGENC * ENC);
    k_convert<<<4096, 256, 0, stream>>>(Wx, wWx, (size_t)NROW * XIN);
    k_convert<<<4096, 256, 0, stream>>>(Ws, wWs, (size_t)NROW * 2 * DD2);
    k_convert<<<8192, 256, 0, stream>>>(Wl, wWl, (size_t)VV * DD2);
    k_embed_x<<<(JJ * BB * EMB + 255) / 256, 256, 0, stream>>>(x, emb, embX);
    hipMemsetAsync(yprev, 0, (size_t)BB * EMB * 2, stream);   // row 0 feedback = zeros

    // encoder: persistent WG per direction, c in VGPRs, h in LDS
    k_encoder<<<2, 1024, 0, stream>>>(embX, wWihF, wWhhF, bF, wWihB, wWhhB, bB, henc);

    // Hpart[j] = h_j * Wx_h^T + b_cell, hoisted out of both scans (2048x2560x1024)
    k_gemm_waves<<<2560, 256, 0, stream>>>(henc, 2 * ENC, wWx, XIN, EMB, b_cell,
                                           Hpart, NROW, JJ * BB / 16, NROW / 16, (2 * ENC) / 32);

    for (int i = 0; i < II; ++i) {
        // Ypart = y_prev * Wx_y^T   (32x2560x512)
        k_gemm_waves<<<40, 256, 0, stream>>>(yprev, EMB, wWx, XIN, 0, nullptr,
                                             Ypart, NROW, 2, NROW / 16, EMB / 32);
        // horizontal scan over J (sequential, on-WGP state)
        k_row<<<1, 1024, 0, stream>>>(Hpart, Ypart, wWs, sJ);
        // logits = s_J * Wl^T + bl  (32x32000x512), written into d_out row i
        float* outRow = out + (size_t)i * BB * VV;
        k_gemm_waves<<<500, 256, 0, stream>>>(sJ, DD2, wWl, DD2, 0, bl,
                                              outRow, VV, 2, VV / 16, DD2 / 32);
        // softmax in place + greedy argmax -> embedding feedback for next row
        k_softmax_feedback<<<BB, 1024, 0, stream>>>(outRow, emb, yprev);
    }
}
